// LIFLayer_65326452572300
// MI455X (gfx1250) — matrix operand order repaired and checked by reading, not compiled, not run on hardware
//
#include <hip/hip_runtime.h>
#include <hip/hip_bf16.h>
#include <math.h>

typedef __attribute__((ext_vector_type(16))) __bf16 v16bf;
typedef __attribute__((ext_vector_type(8)))  __bf16 v8bf;
typedef __attribute__((ext_vector_type(4)))  __bf16 v4bf;
typedef __attribute__((ext_vector_type(8)))  float  v8f;

#define T_DIM 16384
#define K_DIM 2048
#define N_DIM 2048

#define BM 128      // block tile M (2 waves)
#define BN 256      // block tile N (4 waves)
#define BK 32       // K chunk = one WMMA K
#define LDS_STRIDE 40   // 32 + 8 pad -> 80B row stride, conflict-free b128 gathers

// ---------------------------------------------------------------------------
// f32 -> bf16 conversion (vectorized 4-wide)
// ---------------------------------------------------------------------------
__global__ void cvt_f32_to_bf16(const float* __restrict__ in,
                                __bf16* __restrict__ out, int n) {
    int i = (blockIdx.x * blockDim.x + threadIdx.x) * 4;
    if (i < n) {
        float4 f = *(const float4*)(in + i);
        v4bf o = { (__bf16)f.x, (__bf16)f.y, (__bf16)f.z, (__bf16)f.w };
        *(v4bf*)(out + i) = o;
    }
}

// ---------------------------------------------------------------------------
// GEMM: I[m][n] = sum_k x[m][k] * W[n][k] + b[n]
// 256 threads = 8 waves in a 2x4 grid; block tile 128x256; each wave a 64x64
// slab (4x4 WMMA tiles). K-chunks double-buffered in LDS; next chunk's global
// loads are issued before the current chunk's 16 WMMAs.
// ---------------------------------------------------------------------------
__global__ __launch_bounds__(256)
void gemm_bf16_wmma(const __bf16* __restrict__ A,   // [T_DIM, K_DIM] (x)
                    const __bf16* __restrict__ B,   // [N_DIM, K_DIM] (W)
                    const float*  __restrict__ bias,
                    float* __restrict__ I) {        // [T_DIM, N_DIM]
    __shared__ __align__(16) __bf16 As[2][BM][LDS_STRIDE];
    __shared__ __align__(16) __bf16 Bs[2][BN][LDS_STRIDE];

    const int tid  = threadIdx.x;
    const int lane = tid & 31;
    const int wave = tid >> 5;          // 0..7
    const int wm   = wave >> 2;         // 0..1
    const int wn   = wave & 3;          // 0..3

    const int m0 = blockIdx.x * BM;
    const int n0 = blockIdx.y * BN;

    const int lrow = lane & 15;
    const int khiA = (lane >> 4) * 8;   // A frag: elems 0..7 at k+khiA, 8..15 at +16
    const int khiB = (lane >> 4) * 8;   // B frag: elems 0..7 at k+khiB, 8..15 at +16

    // Staging assignment: A -> thread (row = tid>>1, 32B half-row);
    //                     B -> thread (row = tid, full 64B row)
    const int arow  = tid >> 1;
    const int ahalf = (tid & 1) * 16;
    const __bf16* gA = A + (size_t)(m0 + arow) * K_DIM + ahalf;
    const __bf16* gB = B + (size_t)(n0 + tid) * K_DIM;

    v8f acc[4][4] = {};

    // Preload chunk 0 into buffer 0
    {
        v8bf a0 = *(const v8bf*)(gA);
        v8bf a1 = *(const v8bf*)(gA + 8);
        v8bf b0 = *(const v8bf*)(gB);
        v8bf b1 = *(const v8bf*)(gB + 8);
        v8bf b2 = *(const v8bf*)(gB + 16);
        v8bf b3 = *(const v8bf*)(gB + 24);
        *(v8bf*)&As[0][arow][ahalf]     = a0;
        *(v8bf*)&As[0][arow][ahalf + 8] = a1;
        *(v8bf*)&Bs[0][tid][0]  = b0;
        *(v8bf*)&Bs[0][tid][8]  = b1;
        *(v8bf*)&Bs[0][tid][16] = b2;
        *(v8bf*)&Bs[0][tid][24] = b3;
    }
    __syncthreads();

    for (int k0 = 0; k0 < K_DIM; k0 += BK) {
        const int cur = (k0 >> 5) & 1;
        const int nxt = cur ^ 1;
        const bool more = (k0 + BK) < K_DIM;

        // Phase 1: issue next chunk's global loads (latency overlapped w/ WMMA)
        v8bf a0, a1, b0, b1, b2, b3;
        if (more) {
            const __bf16* pa = gA + k0 + BK;
            const __bf16* pb = gB + k0 + BK;
            a0 = *(const v8bf*)(pa);
            a1 = *(const v8bf*)(pa + 8);
            b0 = *(const v8bf*)(pb);
            b1 = *(const v8bf*)(pb + 8);
            b2 = *(const v8bf*)(pb + 16);
            b3 = *(const v8bf*)(pb + 24);
        }

        // Phase 2: fragment gather from LDS + 16 WMMAs
        v16bf afrag[4], bfrag[4];
        #pragma unroll
        for (int i = 0; i < 4; ++i) {
            const __bf16* p = &As[cur][wm * 64 + i * 16 + lrow][0];
            v8bf lo = *(const v8bf*)(p + khiA);
            v8bf hi = *(const v8bf*)(p + khiA + 16);
            afrag[i] = __builtin_shufflevector(lo, hi,
                0,1,2,3,4,5,6,7, 8,9,10,11,12,13,14,15);
        }
        #pragma unroll
        for (int j = 0; j < 4; ++j) {
            const __bf16* p = &Bs[cur][wn * 64 + j * 16 + lrow][0];
            v8bf lo = *(const v8bf*)(p + khiB);
            v8bf hi = *(const v8bf*)(p + khiB + 16);
            bfrag[j] = __builtin_shufflevector(lo, hi,
                0,1,2,3,4,5,6,7, 8,9,10,11,12,13,14,15);
        }
        #pragma unroll
        for (int i = 0; i < 4; ++i)
            #pragma unroll
            for (int j = 0; j < 4; ++j)
                acc[i][j] = __builtin_amdgcn_wmma_f32_16x16x32_bf16(
                    false, afrag[i], false, bfrag[j],
                    (short)0, acc[i][j], false, false);

        // Phase 3: commit staged registers into the other LDS buffer
        if (more) {
            *(v8bf*)&As[nxt][arow][ahalf]     = a0;
            *(v8bf*)&As[nxt][arow][ahalf + 8] = a1;
            *(v8bf*)&Bs[nxt][tid][0]  = b0;
            *(v8bf*)&Bs[nxt][tid][8]  = b1;
            *(v8bf*)&Bs[nxt][tid][16] = b2;
            *(v8bf*)&Bs[nxt][tid][24] = b3;
        }
        __syncthreads();
    }

    // Epilogue: C/D layout -> lane holds column n = (lane&15); VGPR r holds
    // row m = r (lanes 0-15) or r+8 (lanes 16-31).
    const int crow = (lane >> 4) * 8;
    const int ccol = lane & 15;
    #pragma unroll
    for (int j = 0; j < 4; ++j) {
        const int n = n0 + wn * 64 + j * 16 + ccol;
        const float bv = bias[n];
        #pragma unroll
        for (int i = 0; i < 4; ++i) {
            const int mbase = m0 + wm * 64 + i * 16 + crow;
            #pragma unroll
            for (int r = 0; r < 8; ++r)
                I[(size_t)(mbase + r) * N_DIM + n] = acc[i][j][r] + bv;
        }
    }
}

// ---------------------------------------------------------------------------
// LIF scan over T. One thread per hidden channel; reads I in place from d_out
// and overwrites with spikes. 16-deep load batching to hide latency.
// ---------------------------------------------------------------------------
__global__ __launch_bounds__(32)
void lif_scan(float* __restrict__ IO,
              const float* __restrict__ alpha_param,
              const float* __restrict__ threshold) {
    const int h = blockIdx.x * blockDim.x + threadIdx.x;   // 0..N_DIM-1
    const float a  = 1.0f / (1.0f + __expf(-alpha_param[h]));
    const float th = threshold[h];
    float v = 0.0f;

    for (int t0 = 0; t0 < T_DIM; t0 += 16) {
        float cur[16];
        #pragma unroll
        for (int u = 0; u < 16; ++u)
            cur[u] = IO[(size_t)(t0 + u) * N_DIM + h];
        #pragma unroll
        for (int u = 0; u < 16; ++u) {
            v = a * v + cur[u];
            float s = (v > th) ? 1.0f : 0.0f;
            cur[u] = s;
            v = v - v * s;       // reset where spiked
        }
        #pragma unroll
        for (int u = 0; u < 16; ++u)
            IO[(size_t)(t0 + u) * N_DIM + h] = cur[u];
    }
}

// ---------------------------------------------------------------------------
extern "C" void kernel_launch(void* const* d_in, const int* in_sizes, int n_in,
                              void* d_out, int out_size, void* d_ws, size_t ws_size,
                              hipStream_t stream) {
    const float* x           = (const float*)d_in[0];
    const float* W           = (const float*)d_in[1];
    const float* b           = (const float*)d_in[2];
    const float* alpha_param = (const float*)d_in[3];
    const float* threshold   = (const float*)d_in[4];
    float* out = (float*)d_out;

    __bf16* xb = (__bf16*)d_ws;                         // T_DIM*K_DIM bf16 (64 MiB)
    __bf16* Wb = xb + (size_t)T_DIM * K_DIM;            // N_DIM*K_DIM bf16 (8 MiB)

    const int nx = T_DIM * K_DIM;
    const int nw = N_DIM * K_DIM;
    cvt_f32_to_bf16<<<nx / 4 / 256, 256, 0, stream>>>(x, xb, nx);
    cvt_f32_to_bf16<<<nw / 4 / 256, 256, 0, stream>>>(W, Wb, nw);

    dim3 grid(T_DIM / BM, N_DIM / BN);                  // 128 x 8 blocks, 8 waves each
    gemm_bf16_wmma<<<grid, 256, 0, stream>>>(xb, Wb, b, out);

    lif_scan<<<N_DIM / 32, 32, 0, stream>>>(out, alpha_param, threshold);
}
